// SeqFilter_26293789786506
// MI455X (gfx1250) — matrix operand order repaired and checked by reading, not compiled, not run on hardware
//
#include <hip/hip_runtime.h>
#include <cmath>

// ---------------- problem constants ----------------
#define TD   100      // time feature dim
#define MD   172      // memory dim
#define NNB  32       // neighbors
#define CCH  272      // concat channels (100+172)
#define TAU_T 0.3f
#define LAM_T 0.01f
#define K3_NB 8       // batch rows per spectral block

typedef __attribute__((ext_vector_type(16))) __bf16 v16bf;
typedef __attribute__((ext_vector_type(8)))  float  v8f;

#define V8F_ZERO {0.f,0.f,0.f,0.f,0.f,0.f,0.f,0.f}

// ---------------- shared-memory layout for the spectral kernel ----------------
struct K3Smem {
    __bf16 Bls[2 * 4 * 11 * 512];   // [r1/i1][ks4][nt11][lane32][e16]  90112 B
    __bf16 Als[2 * 2 * 4 * 512];    // [re/im][mt2][ks4][lane32][e16]   16384 B
    float  Z[2 * 32 * 176];         // nonlinearized o_r/o_i            45056 B
    float  Cf[8 * 32];              // IFFT+conv-window fold (real)
    float  Sf[8 * 32];              // IFFT+conv-window fold (imag)
    float  sBias[2 * 176];          // rb1 / ib1
    float  ctab[32];                // cos(2*pi*k/32)
    unsigned int smMask[32];        // energy mask per frequency
    float  sGm;                     // mean energy + 1e-6
};

// ============================================================================
// K_prep: build bf16 B-fragment-swizzled folded conv weights Wfs[ck=1376, o=176]
//         Wfs[((ks*11+nt)*32+lane)*16+e] = conv_w[o*1376 + kk]
// ============================================================================
__global__ void k_prep(const float* __restrict__ conv_w, __bf16* __restrict__ Wfs)
{
    int idx = blockIdx.x * 256 + threadIdx.x;
    const int TOT = 43 * 11 * 512;
    if (idx >= TOT) return;
    int e    = idx & 15;
    int lane = (idx >> 4) & 31;
    int rest = idx >> 9;
    int nt   = rest % 11;
    int ks   = rest / 11;                 // < 43
    int kk   = ks * 32 + ((lane >> 4) << 4) + e;   // < 1376
    int col  = nt * 16 + (lane & 15);
    float v  = (col < MD) ? conv_w[(size_t)col * 1376 + kk] : 0.f;
    Wfs[idx] = (__bf16)v;
}

// ============================================================================
// K_dft: per b: tfeat = mask * cos(dt*freq); ortho DFT along neighbor axis;
//        store Y (bf16) + per-frequency energy + per-b energy sum.
// ============================================================================
__global__ __launch_bounds__(256) void k_dft(
    const int*   __restrict__ nbr_ids,
    const float* __restrict__ t_interact,
    const float* __restrict__ nbr_times,
    __bf16*      __restrict__ Yg,        // [B][2][32][100] bf16
    float*       __restrict__ energy,    // [B][32]
    float*       __restrict__ esum)      // [B]
{
    __shared__ float tf[NNB][TD];
    __shared__ float freqs[TD];
    __shared__ float ctab[32];
    __shared__ float red[8][32];
    __shared__ float s_nt[NNB];
    __shared__ int   s_id[NNB];

    const int b = blockIdx.x;
    const int t = threadIdx.x;

    if (t < TD)  freqs[t] = exp10f(-9.0f * (float)t / 99.0f);
    if (t < 32)  ctab[t]  = cosf((float)t * 0.19634954084936207f); // pi/16
    if (t < NNB) { s_nt[t] = nbr_times[b * NNB + t]; s_id[t] = nbr_ids[b * NNB + t]; }
    const float tb = t_interact[b];
    __syncthreads();

    for (int idx = t; idx < NNB * TD; idx += 256) {
        int n = idx / TD, j = idx % TD;
        float dt = tb - s_nt[n];
        tf[n][j] = (s_id[n] == 0) ? 0.0f : cosf(dt * freqs[j]);
    }
    __syncthreads();

    const float s32 = 0.17677669529663687f;  // 1/sqrt(32) (ortho)
    const int f = t & 31;
    const int g = t >> 5;                    // 8 groups share the 100 channels
    float e_acc = 0.f;
    for (int j = g; j < TD; j += 8) {
        float yr = 0.f, yi = 0.f;
        #pragma unroll 8
        for (int n = 0; n < NNB; ++n) {
            float w = tf[n][j];
            int k = (f * n) & 31;
            yr += w * ctab[k];               // cos(2*pi*f*n/32)
            yi -= w * ctab[(k + 24) & 31];   // -sin(2*pi*f*n/32)
        }
        yr *= s32; yi *= s32;
        size_t base = ((size_t)(b * 2 + 0) * NNB + f) * TD + j;
        Yg[base]                     = (__bf16)yr;
        Yg[base + (size_t)NNB * TD]  = (__bf16)yi;
        e_acc += yr * yr + yi * yi;
    }
    red[g][f] = e_acc;
    __syncthreads();
    if (t == 0) {
        float tot = 0.f;
        for (int ff = 0; ff < 32; ++ff) {
            float e = 0.f;
            for (int gg = 0; gg < 8; ++gg) e += red[gg][ff];
            energy[b * NNB + ff] = e;
            tot += e;
        }
        esum[b] = tot;   // deterministic (no atomics)
    }
}

// ============================================================================
// K_reduce: deterministic fixed-order reduction of per-b energy sums
// ============================================================================
__global__ void k_reduce(const float* __restrict__ esum, float* __restrict__ gsum, int B)
{
    __shared__ float red[256];
    int t = threadIdx.x;
    float s = 0.f;
    for (int i = t; i < B; i += 256) s += esum[i];
    red[t] = s;
    __syncthreads();
    if (t == 0) {
        float tot = 0.f;
        for (int i = 0; i < 256; ++i) tot += red[i];
        gsum[0] = tot;
    }
}

// ============================================================================
// K_root: folded rhythm-conv + lin GEMV + double layernorm  -> root[B][172]
// ============================================================================
__global__ __launch_bounds__(256) void k_root(
    const int*   __restrict__ node_ids,
    const float* __restrict__ t_interact,
    const float* __restrict__ node_mem,
    const float* __restrict__ rhythm_w,
    const float* __restrict__ lin_w,
    const float* __restrict__ lin_b,
    const float* __restrict__ mbw, const float* __restrict__ mbb,
    const float* __restrict__ lnw, const float* __restrict__ lnb,
    float* __restrict__ root)
{
    __shared__ float xg[CCH];
    __shared__ float h[MD];
    __shared__ float red[32], red2[32];
    __shared__ float s_mu, s_rs;

    const int b = blockIdx.x;
    const int t = threadIdx.x;
    const float tb = t_interact[b];
    const int nid = node_ids[b];

    for (int c = t; c < CCH; c += 256) {
        float gcl = 0.5f * (rhythm_w[c * 4 + 1] + rhythm_w[c * 4 + 2]);
        float x;
        if (c < TD) x = cosf(tb * exp10f(-9.0f * (float)c / 99.0f));
        else        x = node_mem[(size_t)nid * MD + (c - TD)];
        xg[c] = x * gcl;
    }
    __syncthreads();

    if (t < MD) {
        float acc = lin_b[t];
        for (int c = 0; c < CCH; ++c) acc += xg[c] * lin_w[c * MD + t];
        h[t] = acc;
    }
    __syncthreads();

    // ---- LN1 (mb_ln) ----
    if (t < 32) {
        float s = 0.f, ss = 0.f;
        for (int d = t; d < MD; d += 32) { float v = h[d]; s += v; ss += v * v; }
        red[t] = s; red2[t] = ss;
    }
    __syncthreads();
    if (t == 0) {
        float s = 0.f, ss = 0.f;
        for (int i = 0; i < 32; ++i) { s += red[i]; ss += red2[i]; }
        float mu = s * (1.0f / MD);
        float var = ss * (1.0f / MD) - mu * mu;
        s_mu = mu; s_rs = rsqrtf(var + 1e-5f);
    }
    __syncthreads();
    if (t < MD) h[t] = (h[t] - s_mu) * s_rs * mbw[t] + mbb[t];
    __syncthreads();

    // ---- LN2 (SeqFilter ln) ----
    if (t < 32) {
        float s = 0.f, ss = 0.f;
        for (int d = t; d < MD; d += 32) { float v = h[d]; s += v; ss += v * v; }
        red[t] = s; red2[t] = ss;
    }
    __syncthreads();
    if (t == 0) {
        float s = 0.f, ss = 0.f;
        for (int i = 0; i < 32; ++i) { s += red[i]; ss += red2[i]; }
        float mu = s * (1.0f / MD);
        float var = ss * (1.0f / MD) - mu * mu;
        s_mu = mu; s_rs = rsqrtf(var + 1e-5f);
    }
    __syncthreads();
    if (t < MD) root[(size_t)b * MD + t] = (h[t] - s_mu) * s_rs * lnw[t] + lnb[t];
}

// ============================================================================
// K_spectral: per b: mask -> complex GEMM [32,100]@[100,172] (bf16 WMMA) ->
//             relu+softshrink -> windowed-IFFT fold S[172,8] -> store S (bf16,
//             pre-swizzled into K4's A-fragment layout).
// ============================================================================
__global__ __launch_bounds__(256) void k_spectral(
    const __bf16* __restrict__ Yg,
    const float*  __restrict__ energy,
    const float*  __restrict__ gsum,
    const float*  __restrict__ r1,
    const float*  __restrict__ i1,
    const float*  __restrict__ rb1,
    const float*  __restrict__ ib1,
    __bf16*       __restrict__ Ssw,
    int B)
{
    extern __shared__ __attribute__((aligned(128))) char smem_raw[];
    K3Smem* sm = (K3Smem*)smem_raw;

    const int t  = threadIdx.x;
    const int wv = t >> 5;
    const int ln = t & 31;

    if (t < 32) sm->ctab[t] = cosf((float)t * 0.19634954084936207f);
    if (t == 0) sm->sGm = gsum[0] * (1.0f / ((float)B * (float)NNB)) + 1e-6f;
    for (int c = t; c < 176; c += 256) {
        sm->sBias[c]       = (c < MD) ? rb1[c] : 0.f;
        sm->sBias[176 + c] = (c < MD) ? ib1[c] : 0.f;
    }
    __syncthreads();

    // weights -> LDS in B-fragment layout with contiguous k, same as A (lanes 0-15: K=e, lanes 16-31: K=16+e)
    for (int idx = t; idx < 2 * 4 * 11 * 512; idx += 256) {
        int e    = idx & 15;
        int lane = (idx >> 4) & 31;
        int rest = idx >> 9;
        int nt   = rest % 11; rest /= 11;
        int ks   = rest & 3;
        int m    = rest >> 2;
        int kk   = ks * 32 + ((lane >> 4) << 4) + e;
        int col  = nt * 16 + (lane & 15);
        float v  = (kk < TD && col < MD) ? (m == 0 ? r1 : i1)[kk * MD + col] : 0.f;
        sm->Bls[idx] = (__bf16)v;
    }
    // Cf/Sf: S[k] = sum_f Zr*Cf[k][f] + Zi*Sf[k][f]
    {
        int k = t >> 5, f = t & 31;
        const float c33 = 1.0f / (33.0f * 5.656854249492381f);
        int nb0 = -1, nb1 = -1, nb2 = -1;
        switch (k) {
            case 0: nb0 = 29; nb1 = 30; nb2 = 31; break;
            case 1: nb0 = 30; nb1 = 31; break;
            case 2: nb0 = 31; break;
            case 5: nb0 = 0;  break;
            case 6: nb0 = 0;  nb1 = 1; break;
            case 7: nb0 = 0;  nb1 = 1; nb2 = 2; break;
            default: break;
        }
        float cs = 0.f, sn = 0.f;
        if (nb0 >= 0) { int kk = (nb0 * f) & 31; cs += sm->ctab[kk]; sn += sm->ctab[(kk + 24) & 31]; }
        if (nb1 >= 0) { int kk = (nb1 * f) & 31; cs += sm->ctab[kk]; sn += sm->ctab[(kk + 24) & 31]; }
        if (nb2 >= 0) { int kk = (nb2 * f) & 31; cs += sm->ctab[kk]; sn += sm->ctab[(kk + 24) & 31]; }
        sm->Cf[k * 32 + f] = c33 * ((f == 0 ? 32.0f : 0.0f) - cs);
        sm->Sf[k * 32 + f] = c33 * sn;
    }
    __syncthreads();

    for (int bi = 0; bi < K3_NB; ++bi) {
        const int b = blockIdx.x * K3_NB + bi;
        if (b >= B) break;

        if (t < NNB)
            sm->smMask[t] = (energy[b * NNB + t] >= TAU_T * sm->sGm) ? 1u : 0u;
        __syncthreads();

        // ---- load masked Y into A-fragment layout (u32 granularity) ----
        const __bf16* Yb = Yg + (size_t)b * (2 * NNB * TD);
        for (int idx = t; idx < 4096; idx += 256) {
            int w    = idx & 7;
            int lane = (idx >> 3) & 31;
            int ks   = (idx >> 8) & 3;
            int mt   = (idx >> 10) & 1;
            int part = (idx >> 11) & 1;
            int half = lane >> 4;
            int e0   = w * 2;
            int K    = 16 * half + e0;
            int j0   = ks * 32 + K;
            int f    = mt * 16 + (lane & 15);
            unsigned int val = 0u;
            if (j0 < TD && sm->smMask[f])
                val = *(const unsigned int*)(Yb + (size_t)(part * NNB + f) * TD + j0);
            ((unsigned int*)sm->Als)[idx] = val;
        }
        __syncthreads();

        // ---- complex GEMM via bf16 WMMA: 22 tile-pairs / 8 waves ----
        for (int tp = wv; tp < 22; tp += 8) {
            int mt  = tp / 11;
            int ntt = tp % 11;
            v8f acc_or = V8F_ZERO, acc_ii = V8F_ZERO, acc_oi = V8F_ZERO;
            #pragma unroll
            for (int ks = 0; ks < 4; ++ks) {
                v16bf ar  = *(const v16bf*)(sm->Als + (((0 * 2 + mt) * 4 + ks) * 32 + ln) * 16);
                v16bf ai  = *(const v16bf*)(sm->Als + (((1 * 2 + mt) * 4 + ks) * 32 + ln) * 16);
                v16bf br  = *(const v16bf*)(sm->Bls + (((0 * 4 + ks) * 11 + ntt) * 32 + ln) * 16);
                v16bf bi2 = *(const v16bf*)(sm->Bls + (((1 * 4 + ks) * 11 + ntt) * 32 + ln) * 16);
                acc_or = __builtin_amdgcn_wmma_f32_16x16x32_bf16(false, ar, false, br,  (short)0, acc_or, false, false);
                acc_ii = __builtin_amdgcn_wmma_f32_16x16x32_bf16(false, ai, false, bi2, (short)0, acc_ii, false, false);
                acc_oi = __builtin_amdgcn_wmma_f32_16x16x32_bf16(false, ai, false, br,  (short)0, acc_oi, false, false);
                acc_oi = __builtin_amdgcn_wmma_f32_16x16x32_bf16(false, ar, false, bi2, (short)0, acc_oi, false, false);
            }
            int col   = ntt * 16 + (ln & 15);
            int rbase = mt * 16 + 8 * (ln >> 4);
            if (col < MD) {
                float br_ = sm->sBias[col];
                float bi_ = sm->sBias[176 + col];
                #pragma unroll
                for (int v = 0; v < 8; ++v) {
                    // softshrink(relu(x)) == max(x - lam, 0)
                    sm->Z[(0 * 32 + rbase + v) * 176 + col] = fmaxf(acc_or[v] - acc_ii[v] + br_ - LAM_T, 0.f);
                    sm->Z[(1 * 32 + rbase + v) * 176 + col] = fmaxf(acc_oi[v]             + bi_ - LAM_T, 0.f);
                }
            }
        }
        __syncthreads();

        // ---- S[b][ch][k]: windowed-mean-of-IFFT fold, store swizzled bf16 ----
        for (int ck = t; ck < 1376; ck += 256) {
            int ch = ck >> 3, k = ck & 7;
            float s = 0.f;
            #pragma unroll 8
            for (int f = 0; f < 32; ++f) {
                s += sm->Z[(0 * 32 + f) * 176 + ch] * sm->Cf[k * 32 + f]
                   + sm->Z[(1 * 32 + f) * 176 + ch] * sm->Sf[k * 32 + f];
            }
            int mblk = b >> 4, rrow = b & 15;
            int ks4 = ck >> 5, jj = ck & 31;
            int half, e;
            half = jj >> 4; e = jj & 15;
            int lane = rrow + 16 * half;
            Ssw[((size_t)(mblk * 43 + ks4) * 32 + lane) * 16 + e] = (__bf16)s;
        }
        __syncthreads();
    }
}

// ============================================================================
// K_final: out = S[B,1376] @ Wf[1376,172] (bf16 WMMA) + conv_b + root
// ============================================================================
__global__ __launch_bounds__(256) void k_final(
    const __bf16* __restrict__ Ssw,
    const __bf16* __restrict__ Wfs,
    const float*  __restrict__ conv_b,
    const float*  __restrict__ root,
    float*        __restrict__ out,
    int B)
{
    const int t  = threadIdx.x;
    const int wv = t >> 5;
    const int ln = t & 31;
    const int mblk = blockIdx.x * 8 + wv;       // 16-row block
    if (mblk * 16 >= B) return;                 // uniform per wave

    v8f acc[11];
    #pragma unroll
    for (int n = 0; n < 11; ++n) acc[n] = (v8f)V8F_ZERO;

    for (int ks = 0; ks < 43; ++ks) {
        v16bf a = *(const v16bf*)(Ssw + ((size_t)(mblk * 43 + ks) * 32 + ln) * 16);
        #pragma unroll
        for (int n = 0; n < 11; ++n) {
            v16bf bb = *(const v16bf*)(Wfs + ((size_t)(ks * 11 + n) * 32 + ln) * 16);
            acc[n] = __builtin_amdgcn_wmma_f32_16x16x32_bf16(false, a, false, bb, (short)0, acc[n], false, false);
        }
    }

    const int colb = ln & 15;
    const int rb2  = 8 * (ln >> 4);
    #pragma unroll
    for (int n = 0; n < 11; ++n) {
        int col = n * 16 + colb;
        if (col < MD) {
            float cb = conv_b[col];
            #pragma unroll
            for (int v = 0; v < 8; ++v) {
                int row = mblk * 16 + rb2 + v;
                out[(size_t)row * MD + col] = acc[n][v] + cb + root[(size_t)row * MD + col];
            }
        }
    }
}

// ============================================================================
extern "C" void kernel_launch(void* const* d_in, const int* in_sizes, int n_in,
                              void* d_out, int out_size, void* d_ws, size_t ws_size,
                              hipStream_t stream)
{
    const int B = in_sizes[0];

    const int*   node_ids   = (const int*)  d_in[0];
    const int*   nbr_ids    = (const int*)  d_in[1];
    const float* t_interact = (const float*)d_in[2];
    const float* nbr_times  = (const float*)d_in[3];
    const float* node_mem   = (const float*)d_in[4];
    const float* r1         = (const float*)d_in[5];
    const float* i1         = (const float*)d_in[6];
    const float* rb1        = (const float*)d_in[7];
    const float* ib1        = (const float*)d_in[8];
    const float* conv_w     = (const float*)d_in[9];
    const float* conv_b     = (const float*)d_in[10];
    const float* rhythm_w   = (const float*)d_in[11];
    const float* lin_w      = (const float*)d_in[12];
    const float* lin_b      = (const float*)d_in[13];
    const float* mbw        = (const float*)d_in[14];
    const float* mbb        = (const float*)d_in[15];
    const float* lnw        = (const float*)d_in[16];
    const float* lnb        = (const float*)d_in[17];
    float* outp = (float*)d_out;

    // ---- workspace carve ----
    size_t off = 0;
    auto carve = [&](size_t bytes) { size_t o = off; off = (off + bytes + 255) & ~(size_t)255; return o; };
    size_t off_gsum   = carve(sizeof(float));
    size_t off_esum   = carve((size_t)B * sizeof(float));
    size_t off_energy = carve((size_t)B * NNB * sizeof(float));
    size_t off_Y      = carve((size_t)B * 2 * NNB * TD * sizeof(__bf16));
    size_t off_S      = carve((size_t)(B / 16) * 43 * 512 * sizeof(__bf16));
    size_t off_W      = carve((size_t)43 * 11 * 512 * sizeof(__bf16));
    size_t off_root   = carve((size_t)B * MD * sizeof(float));
    (void)ws_size;

    char* ws = (char*)d_ws;
    float*  gsum   = (float*)(ws + off_gsum);
    float*  esum   = (float*)(ws + off_esum);
    float*  energy = (float*)(ws + off_energy);
    __bf16* Yg     = (__bf16*)(ws + off_Y);
    __bf16* Ssw    = (__bf16*)(ws + off_S);
    __bf16* Wfs    = (__bf16*)(ws + off_W);
    float*  root   = (float*)(ws + off_root);

    // 1) fold + swizzle conv weights into WMMA B-fragment layout
    k_prep<<<(43 * 11 * 512 + 255) / 256, 256, 0, stream>>>(conv_w, Wfs);

    // 2) time-encode + ortho DFT + per-frequency energy
    k_dft<<<B, 256, 0, stream>>>(nbr_ids, t_interact, nbr_times, Yg, energy, esum);

    // 3) deterministic global energy mean
    k_reduce<<<1, 256, 0, stream>>>(esum, gsum, B);

    // 4) root branch (folded rhythm conv + GEMV + double LN)
    k_root<<<B, 256, 0, stream>>>(node_ids, t_interact, node_mem, rhythm_w,
                                  lin_w, lin_b, mbw, mbb, lnw, lnb, root);

    // 5) spectral filter: masked complex GEMM (WMMA) + nonlinearity + IFFT/conv fold
    const size_t smem3 = sizeof(K3Smem);
    (void)hipFuncSetAttribute((const void*)k_spectral,
                              hipFuncAttributeMaxDynamicSharedMemorySize, (int)smem3);
    k_spectral<<<(B + K3_NB - 1) / K3_NB, 256, smem3, stream>>>(
        Yg, energy, gsum, r1, i1, rb1, ib1, Ssw, B);

    // 6) final GEMM (WMMA) + bias + root add
    k_final<<<(B + 127) / 128, 256, 0, stream>>>(Ssw, Wfs, conv_b, root, outp, B);
}
